// SOMNetwork_19275813224498
// MI455X (gfx1250) — compile-verified, matrix operand order checked by reading, not run on hardware
//
#include <hip/hip_runtime.h>
#include <math.h>

typedef __attribute__((ext_vector_type(16))) _Float16 v16h;
typedef __attribute__((ext_vector_type(8)))  _Float16 v8h;
typedef __attribute__((ext_vector_type(8)))  float    v8f;
typedef int v4i_g __attribute__((__vector_size__(16)));   // matches builtin param type

// CDNA5 async global->LDS copy (ASYNCcnt-tracked) if the toolchain exposes it.
#if __has_builtin(__builtin_amdgcn_global_load_async_to_lds_b128) && \
    __has_builtin(__builtin_amdgcn_s_wait_asynccnt)
#define HAVE_ASYNC_LDS 1
#else
#define HAVE_ASYNC_LDS 0
#endif

#if HAVE_ASYNC_LDS
__device__ __forceinline__ void async_cp16(const _Float16* g, _Float16* l) {
    __builtin_amdgcn_global_load_async_to_lds_b128(
        (__attribute__((address_space(1))) v4i_g*)g,
        (__attribute__((address_space(3))) v4i_g*)l, 0, 0);
}
#endif

// ---------------------------------------------------------------- utilities

__global__ void k_zero(float* p, int n) {
    int i = blockIdx.x * blockDim.x + threadIdx.x;
    if (i < n) p[i] = 0.f;
}

// out[r] = sum_k A[r,k]^2   (|w|^2 rows and |x|^2 rows, f32 for accuracy)
__global__ void k_rownorm(const float* __restrict__ A, float* __restrict__ out,
                          int rows, int K) {
    int r = blockIdx.x * blockDim.x + threadIdx.x;
    if (r >= rows) return;
    const float* a = A + (size_t)r * K;
    float s = 0.f;
    for (int k = 0; k < K; ++k) { float v = a[k]; s += v * v; }
    out[r] = s;
}

// f32 (rows,K) -> zero-padded f16 (rowsPad,Kpad); one thread per padded elem.
__global__ void k_pack_half(const float* __restrict__ A, _Float16* __restrict__ out,
                            int rows, int K, int rowsPad, int Kpad) {
    int idx = blockIdx.x * blockDim.x + threadIdx.x;
    int total = rowsPad * Kpad;
    if (idx >= total) return;
    int k = idx % Kpad, r = idx / Kpad;
    float v = (r < rows && k < K) ? A[(size_t)r * K + k] : 0.f;
    out[idx] = (_Float16)v;
}

// ---------------------------------------------------------------- layer 1
__global__ void k_layer1(const float* __restrict__ x,        // (N,3,25,25)
                         const float* __restrict__ rgbw,     // (36,3)
                         const float* __restrict__ wgray,    // (64,25)
                         const float* __restrict__ wn_gray,  // (64)
                         float* __restrict__ dist_rgb,       // (N,6,6,36)
                         float* __restrict__ dist_gray,      // (N,6,6,64)
                         float* __restrict__ stats, int N) {
    __shared__ float s_wg[64 * 25];
    __shared__ float s_rw[36 * 3];
    __shared__ float s_red[256];
    int tid = threadIdx.x;
    for (int i = tid; i < 1600; i += 256) s_wg[i] = wgray[i];
    for (int i = tid; i < 108;  i += 256) s_rw[i] = rgbw[i];
    __syncthreads();

    int pos = blockIdx.x;          // 0..35 -> (i,j)
    int i = pos / 6, j = pos % 6;
    int n = blockIdx.y * 256 + tid;

    float a_rs = 0.f, a_rq = 0.f, a_gs = 0.f, a_gq = 0.f;
    if (n < N) {
        const float* xb = x + (size_t)n * 1875;
        float g[25];
        float s1[3] = {0.f, 0.f, 0.f}, s2c[3] = {0.f, 0.f, 0.f};
        for (int u = 0; u < 5; ++u)
            for (int v = 0; v < 5; ++v) {
                int off = (4 * i + u) * 25 + (4 * j + v);
                float pr = xb[off], pg = xb[625 + off], pb = xb[1250 + off];
                s1[0] += pr; s2c[0] += pr * pr;
                s1[1] += pg; s2c[1] += pg * pg;
                s1[2] += pb; s2c[2] += pb * pb;
                g[u * 5 + v] = 0.2989f * pr + 0.587f * pg + 0.114f * pb;
            }
        float* dr = dist_rgb + ((size_t)n * 36 + pos) * 36;
        for (int o = 0; o < 36; ++o) {
            float d = 0.f;
            for (int c = 0; c < 3; ++c) {
                float wt = s_rw[o * 3 + c];
                float d2 = s2c[c] - 2.f * s1[c] * wt + 25.f * wt * wt;
                d += sqrtf(fmaxf(d2, 1e-12f));
            }
            dr[o] = d;
            a_rs += d; a_rq += d * d;
        }
        float pn = 0.f;
        for (int d0 = 0; d0 < 25; ++d0) pn += g[d0] * g[d0];
        float* dg = dist_gray + ((size_t)n * 36 + pos) * 64;
        for (int o = 0; o < 64; ++o) {
            float dot = 0.f;
            const float* wr = &s_wg[o * 25];
            for (int d0 = 0; d0 < 25; ++d0) dot += g[d0] * wr[d0];
            float d2 = pn + wn_gray[o] - 2.f * dot;
            float dd = sqrtf(fmaxf(d2, 1e-12f));
            dg[o] = dd;
            a_gs += dd; a_gq += dd * dd;
        }
    }
    float vals[4] = {a_rs, a_rq, a_gs, a_gq};
    int   dsts[4] = {2 * pos, 2 * pos + 1, 72 + 2 * pos, 72 + 2 * pos + 1};
    for (int t = 0; t < 4; ++t) {
        s_red[tid] = vals[t];
        __syncthreads();
        for (int s = 128; s > 0; s >>= 1) {
            if (tid < s) s_red[tid] += s_red[tid + s];
            __syncthreads();
        }
        if (tid == 0) atomicAdd(&stats[dsts[t]], s_red[0]);
        __syncthreads();
    }
}

__global__ void k_std1(const float* __restrict__ stats, float* __restrict__ std1, int N) {
    int i = threadIdx.x;
    if (i < 72) {
        float cnt  = (i < 36) ? (float)N * 36.f : (float)N * 64.f;
        int   base = (i < 36) ? 2 * i : 72 + 2 * (i - 36);
        float s = stats[base], q = stats[base + 1];
        float var = (q - s * s / cnt) / (cnt - 1.f);
        std1[i] = sqrtf(fmaxf(var, 0.f));
    }
}

// gauss + crelu(0.4) + sfm (6,6)/(2,2) -> X2 row-major (9N, 100)
__global__ void k_pool1(const float* __restrict__ dist_rgb,
                        const float* __restrict__ dist_gray,
                        const float* __restrict__ std1,
                        float* __restrict__ X2, int N) {
    int idx = blockIdx.x * blockDim.x + threadIdx.x;
    int total = N * 9 * 100;
    if (idx >= total) return;
    int c  = idx % 100;
    int m2 = idx / 100;                  // (n*3+p)*3+q
    int q = m2 % 3, np = m2 / 3;
    int p = np % 3, n = np / 3;
    const float coef[4] = {0.729f, 0.81f, 0.9f, 1.0f};  // alpha^{3,2,1,0}
    float acc = 0.f;
    for (int u = 0; u < 2; ++u)
        for (int v = 0; v < 2; ++v) {
            int h = 2 * p + u, w = 2 * q + v;
            int pos = h * 6 + w;
            float d, sd;
            if (c < 36) { d = dist_rgb [((size_t)n * 36 + pos) * 36 + c];        sd = std1[pos]; }
            else        { d = dist_gray[((size_t)n * 36 + pos) * 64 + (c - 36)]; sd = std1[36 + pos]; }
            float gv = expf(-(d * d) / (2.f * sd * sd + 1e-30f));
            float val = (gv >= 0.4f) ? gv : 0.f;
            acc += coef[u * 2 + v] * val;
        }
    X2[idx] = 0.25f * acc;
}

// --------------------------------------------- WMMA fragment loaders
// Lane L holds row (L&15); K pattern: e=0..7 -> k0 + 8*(L>>4) + e, e=8..15 -> +16.
__device__ __forceinline__ v16h load_frag_g(const _Float16* __restrict__ p) {
    v8h lo = *(const v8h*)p;
    v8h hv = *(const v8h*)(p + 16);
    return __builtin_shufflevector(lo, hv, 0, 1, 2, 3, 4, 5, 6, 7,
                                           8, 9, 10, 11, 12, 13, 14, 15);
}
__device__ __forceinline__ v16h load_frag_l(const _Float16* p) {   // LDS
    v8h lo = *(const v8h*)p;
    v8h hv = *(const v8h*)(p + 16);
    return __builtin_shufflevector(lo, hv, 0, 1, 2, 3, 4, 5, 6, 7,
                                           8, 9, 10, 11, 12, 13, 14, 15);
}

// Shared GEMM core: 128 threads = 4 waves; each wave one 16-row M tile; all
// 4 waves share one 64-wide O block whose W k-chunk (64x32 f16 = 4KB) is
// double-buffered in LDS (async global->LDS when available). A rows are
// streamed from global with prefetch. acc[4] per wave = 16x64 f32 tile.
#define GEMM_CORE(Xh, Wh, Kpad, sW, acc)                                        \
    int tid  = threadIdx.x;                                                     \
    int lane = tid & 31;                                                        \
    int wave = tid >> 5;                                                        \
    int m0 = blockIdx.x * 64 + wave * 16;                                       \
    int ob = blockIdx.y * 64;                                                   \
    int row = lane & 15;                                                        \
    int hi  = lane >> 4;                                                        \
    const _Float16* xp = Xh + (size_t)(m0 + row) * Kpad + hi * 8;               \
    int c0 = tid, c1 = tid + 128;                                               \
    const _Float16* wsrc0 = Wh + (size_t)(ob + (c0 >> 2)) * Kpad + (c0 & 3) * 8;\
    const _Float16* wsrc1 = Wh + (size_t)(ob + (c1 >> 2)) * Kpad + (c1 & 3) * 8;\
    _Float16* wd0[2] = {&sW[0][(c0 >> 2) * 32 + (c0 & 3) * 8],                  \
                        &sW[1][(c0 >> 2) * 32 + (c0 & 3) * 8]};                 \
    _Float16* wd1[2] = {&sW[0][(c1 >> 2) * 32 + (c1 & 3) * 8],                  \
                        &sW[1][(c1 >> 2) * 32 + (c1 & 3) * 8]};                 \
    _Pragma("unroll")                                                           \
    for (int t = 0; t < 4; ++t) acc[t] = v8f{};                                 \
    GEMM_STAGE0;                                                                \
    __syncthreads();                                                            \
    for (int k0 = 0; k0 < Kpad; k0 += 32) {                                     \
        int cur = (k0 >> 5) & 1;                                                \
        bool have = (k0 + 32) < Kpad;                                           \
        GEMM_ISSUE_NEXT;                                                        \
        if (have) __builtin_prefetch(xp + k0 + 64, 0, 0);                       \
        v16h a = load_frag_g(xp + k0);                                          \
        _Pragma("unroll")                                                       \
        for (int t = 0; t < 4; ++t) {                                           \
            const _Float16* bp = &sW[cur][(16 * t + row) * 32 + hi * 8];        \
            v16h b = load_frag_l(bp);                                           \
            acc[t] = __builtin_amdgcn_wmma_f32_16x16x32_f16(                    \
                false, a, false, b, (short)0, acc[t], false, false);            \
        }                                                                       \
        GEMM_FINISH_NEXT;                                                       \
    }

#if HAVE_ASYNC_LDS
#define GEMM_STAGE0                                                             \
    async_cp16(wsrc0, wd0[0]);                                                  \
    async_cp16(wsrc1, wd1[0]);                                                  \
    __builtin_amdgcn_s_wait_asynccnt(0)
#define GEMM_ISSUE_NEXT                                                         \
    if (have) { async_cp16(wsrc0 + k0 + 32, wd0[cur ^ 1]);                      \
                async_cp16(wsrc1 + k0 + 32, wd1[cur ^ 1]); }
#define GEMM_FINISH_NEXT                                                        \
    if (have) __builtin_amdgcn_s_wait_asynccnt(0);                              \
    __syncthreads()
#else
#define GEMM_STAGE0                                                             \
    { v8h r0 = *(const v8h*)wsrc0, r1 = *(const v8h*)wsrc1;                     \
      *(v8h*)wd0[0] = r0; *(v8h*)wd1[0] = r1; }
#define GEMM_ISSUE_NEXT                                                         \
    v8h nr0 = {}, nr1 = {};                                                     \
    if (have) { nr0 = *(const v8h*)(wsrc0 + k0 + 32);                           \
                nr1 = *(const v8h*)(wsrc1 + k0 + 32); }
#define GEMM_FINISH_NEXT                                                        \
    __syncthreads();                                                            \
    if (have) { *(v8h*)wd0[cur ^ 1] = nr0; *(v8h*)wd1[cur ^ 1] = nr1; }         \
    __syncthreads()
#endif

// --------------------------------------------- fused WMMA GEMM + RBF epilogue
__global__ void k_wmma_dist(const _Float16* __restrict__ Xh, const _Float16* __restrict__ Wh,
                            const float* __restrict__ xn, const float* __restrict__ wn,
                            float* __restrict__ dist, int M, int O, int Kpad,
                            float* __restrict__ st) {
    __shared__ __align__(16) _Float16 sW[2][64 * 32];
    v8f acc[4];
    GEMM_CORE(Xh, Wh, Kpad, sW, acc)
    float lsum = 0.f, lsq = 0.f;
#pragma unroll
    for (int t = 0; t < 4; ++t) {
        int ocol = ob + 16 * t + (lane & 15);
        if (ocol < O) {
            float wno = wn[ocol];
#pragma unroll
            for (int r = 0; r < 8; ++r) {
                int m = m0 + r + 8 * hi;     // D layout: M = r + 8*(lane>=16)
                if (m < M) {
                    float d2 = xn[m] + wno - 2.f * acc[t][r];
                    float d = sqrtf(fmaxf(d2, 1e-12f));
                    dist[(size_t)m * O + ocol] = d;
                    lsum += d; lsq += d * d;
                }
            }
        }
    }
#pragma unroll
    for (int off = 16; off >= 1; off >>= 1) {
        lsum += __shfl_xor(lsum, off, 32);
        lsq  += __shfl_xor(lsq,  off, 32);
    }
    if (lane == 0) { atomicAdd(&st[0], lsum); atomicAdd(&st[1], lsq); }
}

// WMMA GEMM + bias epilogue for the FC layer
__global__ void k_wmma_fc(const _Float16* __restrict__ Xh, const _Float16* __restrict__ Wh,
                          const float* __restrict__ bias, float* __restrict__ out,
                          int M, int O, int Kpad) {
    __shared__ __align__(16) _Float16 sW[2][64 * 32];
    v8f acc[4];
    GEMM_CORE(Xh, Wh, Kpad, sW, acc)
#pragma unroll
    for (int t = 0; t < 4; ++t) {
        int ocol = ob + 16 * t + (lane & 15);
        if (ocol < O) {
            float bo = bias[ocol];
#pragma unroll
            for (int r = 0; r < 8; ++r) {
                int m = m0 + r + 8 * hi;
                if (m < M) out[(size_t)m * O + ocol] = acc[t][r] + bo;
            }
        }
    }
}

// ----------------------------------------------- gauss/crelu/pool epilogues
__global__ void k_pool2(const float* __restrict__ d2, const float* __restrict__ st,
                        float* __restrict__ X3, int N) {
    int idx = blockIdx.x * blockDim.x + threadIdx.x;
    int total = 3 * N * 225;
    if (idx >= total) return;
    int c = idx % 225, m3 = idx / 225;      // m3 = n*3+p
    float cnt = (float)(9 * N) * 225.f;
    float s = st[0], qq = st[1];
    float sd = sqrtf(fmaxf((qq - s * s / cnt) / (cnt - 1.f), 0.f));
    float inv = 1.f / (2.f * sd * sd + 1e-30f);
    const float w[3] = {0.81f, 0.9f, 1.0f};  // alpha^(2-q)
    float acc = 0.f;
    for (int q = 0; q < 3; ++q) {
        float d = d2[((size_t)m3 * 3 + q) * 225 + c];
        float gv = expf(-d * d * inv);
        acc += w[q] * ((gv >= 0.1f) ? gv : 0.f);
    }
    X3[idx] = acc * (1.f / 3.f);
}

__global__ void k_pool3(const float* __restrict__ d3, const float* __restrict__ st,
                        float* __restrict__ X4, int N) {
    int idx = blockIdx.x * blockDim.x + threadIdx.x;
    int total = N * 625;
    if (idx >= total) return;
    int c = idx % 625, n = idx / 625;
    float cnt = (float)(3 * N) * 625.f;
    float s = st[0], qq = st[1];
    float sd = sqrtf(fmaxf((qq - s * s / cnt) / (cnt - 1.f), 0.f));
    float inv = 1.f / (2.f * sd * sd + 1e-30f);
    const float w[3] = {0.81f, 0.9f, 1.0f};  // alpha^(2-p)
    float acc = 0.f;
    for (int p = 0; p < 3; ++p) {
        float d = d3[((size_t)n * 3 + p) * 625 + c];
        float gv = expf(-d * d * inv);
        acc += w[p] * ((gv >= 0.01f) ? gv : 0.f);
    }
    X4[idx] = acc * (1.f / 3.f);
}

__global__ void k_final4(float* __restrict__ d4, const float* __restrict__ st, int N) {
    int idx = blockIdx.x * blockDim.x + threadIdx.x;
    int total = N * 1225;
    if (idx >= total) return;
    float cnt = (float)N * 1225.f;
    float s = st[0], qq = st[1];
    float sd = sqrtf(fmaxf((qq - s * s / cnt) / (cnt - 1.f), 0.f));
    float inv = 1.f / (2.f * sd * sd + 1e-30f);
    float d = d4[idx];
    float gv = expf(-d * d * inv);
    d4[idx] = (gv >= 0.01f) ? gv : 0.f;     // in place: becomes FC input
}

__global__ void k_softmax(const float* __restrict__ logits, float* __restrict__ out, int O) {
    __shared__ float red[256];
    int row = blockIdx.x, tid = threadIdx.x;
    const float* lr = logits + (size_t)row * O;
    float* orow = out + (size_t)row * O;
    float mx = -1e30f;
    for (int i = tid; i < O; i += 256) mx = fmaxf(mx, lr[i]);
    red[tid] = mx; __syncthreads();
    for (int s = 128; s > 0; s >>= 1) {
        if (tid < s) red[tid] = fmaxf(red[tid], red[tid + s]);
        __syncthreads();
    }
    mx = red[0]; __syncthreads();
    float sum = 0.f;
    for (int i = tid; i < O; i += 256) sum += expf(lr[i] - mx);
    red[tid] = sum; __syncthreads();
    for (int s = 128; s > 0; s >>= 1) {
        if (tid < s) red[tid] += red[tid + s];
        __syncthreads();
    }
    float inv = 1.f / red[0];
    for (int i = tid; i < O; i += 256) orow[i] = expf(lr[i] - mx) * inv;
}

// ---------------------------------------------------------------- launcher
extern "C" void kernel_launch(void* const* d_in, const int* in_sizes, int n_in,
                              void* d_out, int out_size, void* d_ws, size_t ws_size,
                              hipStream_t stream) {
    (void)n_in; (void)out_size; (void)ws_size;
    const float* x    = (const float*)d_in[0];
    const float* rgbw = (const float*)d_in[1];
    const float* wg   = (const float*)d_in[2];
    const float* w2   = (const float*)d_in[3];
    const float* w3   = (const float*)d_in[4];
    const float* w4   = (const float*)d_in[5];
    const float* fcw  = (const float*)d_in[6];
    const float* fcb  = (const float*)d_in[7];
    float* out = (float*)d_out;
    const int N = in_sizes[0] / 1875;          // (N,3,25,25)

    // GEMM geometry: K padded to x32, W rows padded to x64, X rows to x64
    const int K2 = 100,  Kp2 = 128,  O2 = 225,  Op2 = 256;
    const int K3 = 225,  Kp3 = 256,  O3 = 625,  Op3 = 640;
    const int K4 = 625,  Kp4 = 640,  O4 = 1225, Op4 = 1280;
    const int Kf = 1225, Kpf = 1248, Of = 1000, Opf = 1024;
    const int M2 = 9 * N, M3 = 3 * N, M4 = N;
    auto pad64 = [](int v) { return (v + 63) & ~63; };
    const int Mp2 = pad64(M2), Mp3 = pad64(M3), Mp4 = pad64(M4);

    float* ws = (float*)d_ws;
    size_t off = 0;
    auto alloc = [&](size_t nfloats) -> float* {
        float* p = ws + off;
        off += (nfloats + 255) & ~(size_t)255;   // 1KB-aligned slabs
        return p;
    };
    auto allocH = [&](size_t nhalf) -> _Float16* {
        return (_Float16*)alloc((nhalf + 1) / 2);
    };
    float* stats = alloc(256);
    float* std1  = alloc(128);
    float* wn_g  = alloc(64);
    float* wn2   = alloc(O2);
    float* wn3   = alloc(O3);
    float* wn4   = alloc(O4);
    float* xn2   = alloc((size_t)M2);
    float* xn3   = alloc((size_t)M3);
    float* xn4   = alloc((size_t)M4);
    float* drgb  = alloc((size_t)N * 36 * 36);
    float* dgray = alloc((size_t)N * 36 * 64);
    float* X2    = alloc((size_t)M2 * K2);
    float* d2b   = alloc((size_t)M2 * O2);
    float* X3    = alloc((size_t)M3 * K3);
    float* d3b   = alloc((size_t)M3 * O3);
    float* X4    = alloc((size_t)M4 * K4);
    float* d4b   = alloc((size_t)M4 * Kf);     // (N,1225), reused as FC input
    float* lg    = alloc((size_t)N * Of);
    _Float16* W2h = allocH((size_t)Op2 * Kp2);
    _Float16* W3h = allocH((size_t)Op3 * Kp3);
    _Float16* W4h = allocH((size_t)Op4 * Kp4);
    _Float16* FWh = allocH((size_t)Opf * Kpf);
    _Float16* X2h = allocH((size_t)Mp2 * Kp2);
    _Float16* X3h = allocH((size_t)Mp3 * Kp3);
    _Float16* X4h = allocH((size_t)Mp4 * Kp4);
    _Float16* X5h = allocH((size_t)Mp4 * Kpf);

    auto cdiv = [](int a, int b) { return (a + b - 1) / b; };

    k_zero<<<1, 256, 0, stream>>>(stats, 256);

    // weight norms + packed f16 weights (tiny, once per call)
    k_rownorm<<<1, 128, 0, stream>>>(wg, wn_g, 64, 25);
    k_rownorm<<<2, 128, 0, stream>>>(w2, wn2, O2, K2);
    k_rownorm<<<5, 128, 0, stream>>>(w3, wn3, O3, K3);
    k_rownorm<<<10, 128, 0, stream>>>(w4, wn4, O4, K4);
    k_pack_half<<<cdiv(Op2 * Kp2, 256), 256, 0, stream>>>(w2,  W2h, O2, K2, Op2, Kp2);
    k_pack_half<<<cdiv(Op3 * Kp3, 256), 256, 0, stream>>>(w3,  W3h, O3, K3, Op3, Kp3);
    k_pack_half<<<cdiv(Op4 * Kp4, 256), 256, 0, stream>>>(w4,  W4h, O4, K4, Op4, Kp4);
    k_pack_half<<<cdiv(Opf * Kpf, 256), 256, 0, stream>>>(fcw, FWh, Of, Kf, Opf, Kpf);

    // layer 1
    dim3 g1(36, cdiv(N, 256));
    k_layer1<<<g1, 256, 0, stream>>>(x, rgbw, wg, wn_g, drgb, dgray, stats, N);
    k_std1<<<1, 128, 0, stream>>>(stats, std1, N);
    k_pool1<<<cdiv(N * 900, 256), 256, 0, stream>>>(drgb, dgray, std1, X2, N);

    // layer 2: (9N x 100) x (100 x 225)
    k_rownorm<<<cdiv(M2, 128), 128, 0, stream>>>(X2, xn2, M2, K2);
    k_pack_half<<<cdiv(Mp2 * Kp2, 256), 256, 0, stream>>>(X2, X2h, M2, K2, Mp2, Kp2);
    dim3 g2(Mp2 / 64, Op2 / 64);
    k_wmma_dist<<<g2, 128, 0, stream>>>(X2h, W2h, xn2, wn2, d2b, M2, O2, Kp2, stats + 144);
    k_pool2<<<cdiv(M3 * O2, 256), 256, 0, stream>>>(d2b, stats + 144, X3, N);

    // layer 3: (3N x 225) x (225 x 625)
    k_rownorm<<<cdiv(M3, 128), 128, 0, stream>>>(X3, xn3, M3, K3);
    k_pack_half<<<cdiv(Mp3 * Kp3, 256), 256, 0, stream>>>(X3, X3h, M3, K3, Mp3, Kp3);
    dim3 g3(Mp3 / 64, Op3 / 64);
    k_wmma_dist<<<g3, 128, 0, stream>>>(X3h, W3h, xn3, wn3, d3b, M3, O3, Kp3, stats + 146);
    k_pool3<<<cdiv(M4 * O3, 256), 256, 0, stream>>>(d3b, stats + 146, X4, N);

    // layer 4: (N x 625) x (625 x 1225)
    k_rownorm<<<cdiv(M4, 128), 128, 0, stream>>>(X4, xn4, M4, K4);
    k_pack_half<<<cdiv(Mp4 * Kp4, 256), 256, 0, stream>>>(X4, X4h, M4, K4, Mp4, Kp4);
    dim3 g4(Mp4 / 64, Op4 / 64);
    k_wmma_dist<<<g4, 128, 0, stream>>>(X4h, W4h, xn4, wn4, d4b, M4, O4, Kp4, stats + 148);
    k_final4<<<cdiv(N * 1225, 256), 256, 0, stream>>>(d4b, stats + 148, N);

    // FC + softmax
    k_pack_half<<<cdiv(Mp4 * Kpf, 256), 256, 0, stream>>>(d4b, X5h, M4, Kf, Mp4, Kpf);
    dim3 g5(Mp4 / 64, Opf / 64);
    k_wmma_fc<<<g5, 128, 0, stream>>>(X5h, FWh, fcb, lg, M4, Of, Kpf);
    k_softmax<<<N, 256, 0, stream>>>(lg, out, Of);
}
